// Narrow_MultiHeadAttention_42125039239946
// MI455X (gfx1250) — compile-verified, matrix operand order checked by reading, not run on hardware
//
#include <hip/hip_runtime.h>

// ---------------------------------------------------------------------------
// Types for CDNA5 WMMA (wave32)
// ---------------------------------------------------------------------------
typedef __bf16 bf16;
typedef __bf16 v16bf __attribute__((ext_vector_type(16)));
typedef __bf16 v8bf  __attribute__((ext_vector_type(8)));
typedef float  v8f   __attribute__((ext_vector_type(8)));
typedef float  v4f   __attribute__((ext_vector_type(4)));

#define WMMA_BF16(a, b, c) \
  __builtin_amdgcn_wmma_f32_16x16x32_bf16(false, (a), false, (b), (short)0, (c), false, false)

// TDM availability / builtin arity detection
#if defined(__has_builtin)
#if __has_builtin(__builtin_amdgcn_tensor_load_to_lds)
#define HAVE_TDM 1
#endif
#endif
#if __has_include(<hip/amd_detail/amd_gfx1250_TDM.h>)
#define TDM_ARITY6 1
#endif

// Problem constants (reference: B=8, S=1024, D=512, H=8)
#define BB 8
#define SS 1024
#define DD 512
#define HH 8
#define NQKV (DD * HH)      // 4096
#define NTOK (BB * SS)      // 8192

// ---------------------------------------------------------------------------
// Fragment helpers.
// 16-bit A-matrix 16x32 layout (ISA 7.12.2): lane L&15 = row M, kb = 8*(L>>4);
// elements 0..7 hold K = kb+0..7, elements 8..15 hold K = kb+16..23.
// B fragments use the identical pattern on the transposed (N x K) operand.
// ---------------------------------------------------------------------------
__device__ __forceinline__ void frag_lane(int lane, int& row, int& kb) {
  row = lane & 15;
  kb  = (lane >> 4) << 3;
}

__device__ __forceinline__ v16bf load_frag(const bf16* __restrict__ mat, int ld,
                                           int m0, int kk, int row, int kb) {
  const bf16* p = mat + (size_t)(m0 + row) * ld + kk + kb;
  v8bf lo = *(const v8bf*)(p);
  v8bf hi = *(const v8bf*)(p + 16);
  v16bf r;
#pragma unroll
  for (int i = 0; i < 8; ++i) { r[i] = lo[i]; r[8 + i] = hi[i]; }
  return r;
}

__device__ __forceinline__ v16bf load_frag(const float* __restrict__ mat, int ld,
                                           int m0, int kk, int row, int kb) {
  const float* p = mat + (size_t)(m0 + row) * ld + kk + kb;
  const v4f* q = (const v4f*)p;
  v4f x0 = q[0], x1 = q[1], x2 = q[4], x3 = q[5];  // floats +0,+4 and +16,+20
  v16bf r;
#pragma unroll
  for (int i = 0; i < 4; ++i) {
    r[i]      = (bf16)x0[i];
    r[4 + i]  = (bf16)x1[i];
    r[8 + i]  = (bf16)x2[i];
    r[12 + i] = (bf16)x3[i];
  }
  return r;
}

// Shared inner loop: 2x2 tiles of 16x16 per wave, K step 32.
template <typename AT>
__device__ __forceinline__ void mma_loop(const AT* __restrict__ A, int lda,
                                         const bf16* __restrict__ Bt, int ldb,
                                         int m_w, int n_w, int Kdim,
                                         int row, int kb, v8f acc[2][2]) {
#pragma unroll 2
  for (int kk = 0; kk < Kdim; kk += 32) {
    v16bf a0 = load_frag(A,  lda, m_w,      kk, row, kb);
    v16bf a1 = load_frag(A,  lda, m_w + 16, kk, row, kb);
    v16bf b0 = load_frag(Bt, ldb, n_w,      kk, row, kb);
    v16bf b1 = load_frag(Bt, ldb, n_w + 16, kk, row, kb);
    acc[0][0] = WMMA_BF16(a0, b0, acc[0][0]);
    acc[0][1] = WMMA_BF16(a0, b1, acc[0][1]);
    acc[1][0] = WMMA_BF16(a1, b0, acc[1][0]);
    acc[1][1] = WMMA_BF16(a1, b1, acc[1][1]);
  }
}

// ---------------------------------------------------------------------------
// Weight transpose + f32 -> bf16 convert:  src (K x N) -> dst (N x K)
// ---------------------------------------------------------------------------
__global__ void k_transpose_bf16(const float* __restrict__ src,
                                 bf16* __restrict__ dst, int K, int N) {
  __shared__ float tile[32][33];
  const int k0 = blockIdx.y * 32;
  const int n0 = blockIdx.x * 32;
  for (int i = threadIdx.y; i < 32; i += 8)
    tile[i][threadIdx.x] = src[(size_t)(k0 + i) * N + n0 + threadIdx.x];
  __syncthreads();
  for (int i = threadIdx.y; i < 32; i += 8)
    dst[(size_t)(n0 + i) * K + k0 + threadIdx.x] = (bf16)tile[threadIdx.x][i];
}

// ---------------------------------------------------------------------------
// QKV projection: dst = split_heads((A @ W + b) * scale)
//   mode 0: dst laid out (B, H, S, DD)   (Q, K)
//   mode 1: dst laid out (B, H, DD, S)   (V, pre-transposed for attn @ V)
// ---------------------------------------------------------------------------
__global__ void k_proj(const float* __restrict__ A,
                       const bf16* __restrict__ Wt,
                       const float* __restrict__ bias,
                       bf16* __restrict__ dst, float scale, int mode) {
  const int lane = threadIdx.x & 31;
  const int wid  = threadIdx.x >> 5;
  int row, kb; frag_lane(lane, row, kb);

  const int m_w = blockIdx.y * 128 + (wid >> 1) * 32;
  const int n_w = blockIdx.x * 64  + (wid & 1) * 32;

  v8f acc[2][2] = {};
  mma_loop(A, DD, Wt, DD, m_w, n_w, DD, row, kb, acc);

  const int cn = lane & 15;
  const int rm = (lane >> 4) << 3;
#pragma unroll
  for (int tm = 0; tm < 2; ++tm)
#pragma unroll
    for (int tn = 0; tn < 2; ++tn) {
      const int n  = n_w + tn * 16 + cn;
      const float bsv = bias[n];
      const int h = n >> 9, d = n & (DD - 1);
#pragma unroll
      for (int r = 0; r < 8; ++r) {
        const int m = m_w + tm * 16 + r + rm;
        const int b = m >> 10, s = m & (SS - 1);
        const float val = (acc[tm][tn][r] + bsv) * scale;
        const size_t off = (mode == 0)
            ? (((size_t)(b * HH + h) * SS + s) * DD + d)
            : (((size_t)(b * HH + h) * DD + d) * SS + s);
        dst[off] = (bf16)val;
      }
    }
}

// ---------------------------------------------------------------------------
// logits = Q @ K^T + mask * -1e9   (1/sqrt(D) already folded into Q)
// ---------------------------------------------------------------------------
__global__ void k_scores(const bf16* __restrict__ Q,
                         const bf16* __restrict__ K,
                         const float* __restrict__ mask,
                         float* __restrict__ logits) {
  const int bh = blockIdx.z;
  const bf16* Qh = Q + (size_t)bh * SS * DD;
  const bf16* Kh = K + (size_t)bh * SS * DD;
  const float* Mb = mask + (size_t)(bh >> 3) * SS * SS;
  float* Lh = logits + (size_t)bh * SS * SS;

  const int lane = threadIdx.x & 31;
  const int wid  = threadIdx.x >> 5;
  int row, kb; frag_lane(lane, row, kb);

  const int m_w = blockIdx.y * 128 + (wid >> 1) * 32;
  const int n_w = blockIdx.x * 64  + (wid & 1) * 32;

  v8f acc[2][2] = {};
  mma_loop(Qh, DD, Kh, DD, m_w, n_w, DD, row, kb, acc);

  const int cn = lane & 15;
  const int rm = (lane >> 4) << 3;
#pragma unroll
  for (int tm = 0; tm < 2; ++tm)
#pragma unroll
    for (int tn = 0; tn < 2; ++tn) {
      const int n = n_w + tn * 16 + cn;
#pragma unroll
      for (int r = 0; r < 8; ++r) {
        const int m = m_w + tm * 16 + r + rm;
        Lh[(size_t)m * SS + n] = acc[tm][tn][r] + Mb[(size_t)m * SS + n] * (-1e9f);
      }
    }
}

// ---------------------------------------------------------------------------
// Row softmax over 1024 cols, in place on f32 attn; emits bf16 copy.
// ---------------------------------------------------------------------------
__device__ __forceinline__ float wave_max(float v) {
#pragma unroll
  for (int o = 16; o > 0; o >>= 1) v = fmaxf(v, __shfl_xor(v, o, 32));
  return v;
}
__device__ __forceinline__ float wave_sum(float v) {
#pragma unroll
  for (int o = 16; o > 0; o >>= 1) v += __shfl_xor(v, o, 32);
  return v;
}

__global__ void k_softmax(float* __restrict__ attn, bf16* __restrict__ attn_bf) {
  __shared__ float red[8];
  const size_t base = (size_t)blockIdx.x * SS;
  const int t = threadIdx.x;
  float x0 = attn[base + t];
  float x1 = attn[base + t + 256];
  float x2 = attn[base + t + 512];
  float x3 = attn[base + t + 768];

  float m = wave_max(fmaxf(fmaxf(x0, x1), fmaxf(x2, x3)));
  if ((t & 31) == 0) red[t >> 5] = m;
  __syncthreads();
  m = fmaxf(fmaxf(fmaxf(red[0], red[1]), fmaxf(red[2], red[3])),
            fmaxf(fmaxf(red[4], red[5]), fmaxf(red[6], red[7])));

  const float e0 = __expf(x0 - m), e1 = __expf(x1 - m);
  const float e2 = __expf(x2 - m), e3 = __expf(x3 - m);
  float s = wave_sum(e0 + e1 + e2 + e3);
  __syncthreads();
  if ((t & 31) == 0) red[t >> 5] = s;
  __syncthreads();
  const float inv = 1.0f / (red[0] + red[1] + red[2] + red[3] +
                            red[4] + red[5] + red[6] + red[7]);

  const float o0 = e0 * inv, o1 = e1 * inv, o2 = e2 * inv, o3 = e3 * inv;
  attn[base + t]       = o0;  attn_bf[base + t]       = (bf16)o0;
  attn[base + t + 256] = o1;  attn_bf[base + t + 256] = (bf16)o1;
  attn[base + t + 512] = o2;  attn_bf[base + t + 512] = (bf16)o2;
  attn[base + t + 768] = o3;  attn_bf[base + t + 768] = (bf16)o3;
}

// ---------------------------------------------------------------------------
// TDM descriptor issue: 2D bf16 tile (tile_d1 rows x tile_d0 elems) from a
// row-major tensor (tensor_d1 x tensor_d0, row stride stride_d0 elems) into
// LDS at byte offset lds_off.  D# layout per CDNA5 ISA sec. 8.3/8.4.
// ---------------------------------------------------------------------------
#if HAVE_TDM
typedef unsigned int u32x4 __attribute__((ext_vector_type(4)));
typedef int i32x4 __attribute__((ext_vector_type(4)));
typedef int i32x8 __attribute__((ext_vector_type(8)));

__device__ __forceinline__ void tdm_load_tile_2d(const bf16* gsrc, unsigned lds_off,
                                                 unsigned tensor_d0, unsigned tensor_d1,
                                                 unsigned stride_d0,
                                                 unsigned tile_d0, unsigned tile_d1) {
  const unsigned long long ga = (unsigned long long)(uintptr_t)gsrc;
  u32x4 g0;
  g0[0] = 1u;                                            // count=1, user desc
  g0[1] = lds_off;                                       // lds_addr (bytes)
  g0[2] = (unsigned)ga;                                  // global_addr[31:0]
  g0[3] = ((unsigned)(ga >> 32) & 0x01ffffffu)           // global_addr[56:32]
          | (2u << 30);                                  // type = 2 ("image")
  i32x8 g1;
  g1[0] = (int)(1u << 16);                               // data_size=1 (2B), mask=0
  g1[1] = (int)((tensor_d0 & 0xffffu) << 16);            // tensor_dim0[15:0]
  g1[2] = (int)((tensor_d0 >> 16) | ((tensor_d1 & 0xffffu) << 16));
  g1[3] = (int)((tensor_d1 >> 16) | ((tile_d0 & 0xffffu) << 16));
  g1[4] = (int)(tile_d1 & 0xffffu);                      // tile_dim1; tile_dim2=0
  g1[5] = (int)stride_d0;                                // tensor_dim0_stride[31:0]
  g1[6] = 0;                                             // stride[47:32], dim1_stride lo
  g1[7] = 0;
  i32x4 z4 = {0, 0, 0, 0};
#if TDM_ARITY6
  i32x8 z8 = {0, 0, 0, 0, 0, 0, 0, 0};
  __builtin_amdgcn_tensor_load_to_lds(g0, g1, z4, z4, z8, 0);
#else
  __builtin_amdgcn_tensor_load_to_lds(g0, g1, z4, z4, 0);
#endif
}
#endif  // HAVE_TDM

// ---------------------------------------------------------------------------
// ctx = attn @ V ; attn_bf (B,H,S,S), Vt (B,H,DD,S); ctx (NTOK, NQKV) bf16
// TDM path: double-buffered LDS tiles (A: 128x32, B: 64x32), wave0 drives the
// Tensor Data Mover, block barriers publish tiles, DMA overlaps WMMA.
// ---------------------------------------------------------------------------
__global__ void k_ctx(const bf16* __restrict__ attn_bf,
                      const bf16* __restrict__ Vt,
                      bf16* __restrict__ ctx) {
  const int bh = blockIdx.z;
  const int b = bh >> 3, h = bh & 7;
  const bf16* Ah = attn_bf + (size_t)bh * SS * SS;   // (SS, SS),  ld = SS
  const bf16* Bh = Vt      + (size_t)bh * DD * SS;   // (DD, SS),  ld = SS

  const int lane = threadIdx.x & 31;
  const int wid  = threadIdx.x >> 5;
  int row, kb; frag_lane(lane, row, kb);

  const int m_wg = blockIdx.y * 128;
  const int n_wg = blockIdx.x * 64;
  const int m_w  = (wid >> 1) * 32;   // wave offsets inside the LDS tiles
  const int n_w  = (wid & 1) * 32;

  v8f acc[2][2] = {};

#if HAVE_TDM
  __shared__ __align__(16) bf16 ldsA[2][128 * 32];
  __shared__ __align__(16) bf16 ldsB[2][64 * 32];
  const int niter = SS / 32;

  if (wid == 0) {
    tdm_load_tile_2d(Ah + (size_t)m_wg * SS, (unsigned)(uintptr_t)&ldsA[0][0],
                     SS, SS, SS, 32, 128);
    tdm_load_tile_2d(Bh + (size_t)n_wg * SS, (unsigned)(uintptr_t)&ldsB[0][0],
                     SS, DD, SS, 32, 64);
  }
  for (int it = 0; it < niter; ++it) {
    const int buf = it & 1;
    if (wid == 0) __builtin_amdgcn_s_wait_tensorcnt(0);
    __syncthreads();                       // LDS tiles for `buf` are visible
    if (wid == 0 && it + 1 < niter) {      // prefetch next k-slab into buf^1
      const int kk = (it + 1) * 32;
      tdm_load_tile_2d(Ah + (size_t)m_wg * SS + kk,
                       (unsigned)(uintptr_t)&ldsA[buf ^ 1][0], SS, SS, SS, 32, 128);
      tdm_load_tile_2d(Bh + (size_t)n_wg * SS + kk,
                       (unsigned)(uintptr_t)&ldsB[buf ^ 1][0], SS, DD, SS, 32, 64);
    }
    const bf16* As = &ldsA[buf][0];
    const bf16* Bs = &ldsB[buf][0];
    v16bf a0 = load_frag(As, 32, m_w,      0, row, kb);
    v16bf a1 = load_frag(As, 32, m_w + 16, 0, row, kb);
    v16bf b0 = load_frag(Bs, 32, n_w,      0, row, kb);
    v16bf b1 = load_frag(Bs, 32, n_w + 16, 0, row, kb);
    acc[0][0] = WMMA_BF16(a0, b0, acc[0][0]);
    acc[0][1] = WMMA_BF16(a0, b1, acc[0][1]);
    acc[1][0] = WMMA_BF16(a1, b0, acc[1][0]);
    acc[1][1] = WMMA_BF16(a1, b1, acc[1][1]);
    __syncthreads();                       // all reads of `buf` done before reuse
  }
#else
  mma_loop(Ah, SS, Bh, SS, m_wg + m_w, n_wg + n_w, SS, row, kb, acc);
#endif

  const int cn = lane & 15;
  const int rm = (lane >> 4) << 3;
#pragma unroll
  for (int tm = 0; tm < 2; ++tm)
#pragma unroll
    for (int tn = 0; tn < 2; ++tn) {
      const int n = n_wg + n_w + tn * 16 + cn;
#pragma unroll
      for (int r = 0; r < 8; ++r) {
        const int m = m_wg + m_w + tm * 16 + r + rm;
        ctx[(size_t)(b * SS + m) * NQKV + h * DD + n] = (bf16)acc[tm][tn][r];
      }
    }
}

// ---------------------------------------------------------------------------
// out = ctx @ Wo + bo  -> f32 out region of d_out
// ---------------------------------------------------------------------------
__global__ void k_out(const bf16* __restrict__ ctx,
                      const bf16* __restrict__ WoT,
                      const float* __restrict__ bo,
                      float* __restrict__ out) {
  const int lane = threadIdx.x & 31;
  const int wid  = threadIdx.x >> 5;
  int row, kb; frag_lane(lane, row, kb);

  const int m_w = blockIdx.y * 128 + (wid >> 1) * 32;
  const int n_w = blockIdx.x * 64  + (wid & 1) * 32;

  v8f acc[2][2] = {};
  mma_loop(ctx, NQKV, WoT, NQKV, m_w, n_w, NQKV, row, kb, acc);

  const int cn = lane & 15;
  const int rm = (lane >> 4) << 3;
#pragma unroll
  for (int tm = 0; tm < 2; ++tm)
#pragma unroll
    for (int tn = 0; tn < 2; ++tn) {
      const int n = n_w + tn * 16 + cn;
      const float bsv = bo[n];
#pragma unroll
      for (int r = 0; r < 8; ++r) {
        const int m = m_w + tm * 16 + r + rm;
        out[(size_t)m * DD + n] = acc[tm][tn][r] + bsv;
      }
    }
}

// ---------------------------------------------------------------------------
// Launch
// ---------------------------------------------------------------------------
extern "C" void kernel_launch(void* const* d_in, const int* in_sizes, int n_in,
                              void* d_out, int out_size, void* d_ws, size_t ws_size,
                              hipStream_t stream) {
  (void)in_sizes; (void)n_in; (void)out_size; (void)ws_size;

  const float* v_ip = (const float*)d_in[0];
  const float* k_ip = (const float*)d_in[1];
  const float* q_ip = (const float*)d_in[2];
  const float* mask = (const float*)d_in[3];
  const float* wq   = (const float*)d_in[4];
  const float* bq   = (const float*)d_in[5];
  const float* wk   = (const float*)d_in[6];
  const float* bk   = (const float*)d_in[7];
  const float* wv   = (const float*)d_in[8];
  const float* bv   = (const float*)d_in[9];
  const float* wo   = (const float*)d_in[10];
  const float* bo   = (const float*)d_in[11];

  char* ws = (char*)d_ws;
  size_t off = 0;
  bf16* wqT = (bf16*)(ws + off); off += (size_t)NQKV * DD * 2;
  bf16* wkT = (bf16*)(ws + off); off += (size_t)NQKV * DD * 2;
  bf16* wvT = (bf16*)(ws + off); off += (size_t)NQKV * DD * 2;
  bf16* woT = (bf16*)(ws + off); off += (size_t)DD * NQKV * 2;
  bf16* Qb  = (bf16*)(ws + off); off += (size_t)BB * HH * SS * DD * 2;
  bf16* Kb  = (bf16*)(ws + off); off += (size_t)BB * HH * SS * DD * 2;
  bf16* Vt  = (bf16*)(ws + off); off += (size_t)BB * HH * DD * SS * 2;
  bf16* Abf = (bf16*)(ws + off); off += (size_t)BB * HH * SS * SS * 2;
  bf16* Ctx = (bf16*)(ws + off); off += (size_t)NTOK * NQKV * 2;

  float* outp = (float*)d_out;
  float* attn = outp + (size_t)NTOK * DD;   // out first, then attn

  const float qscale = 0.044194173824159216f;  // 1/sqrt(512)

  dim3 tb(32, 8, 1);
  k_transpose_bf16<<<dim3(NQKV / 32, DD / 32, 1), tb, 0, stream>>>(wq, wqT, DD, NQKV);
  k_transpose_bf16<<<dim3(NQKV / 32, DD / 32, 1), tb, 0, stream>>>(wk, wkT, DD, NQKV);
  k_transpose_bf16<<<dim3(NQKV / 32, DD / 32, 1), tb, 0, stream>>>(wv, wvT, DD, NQKV);
  k_transpose_bf16<<<dim3(DD / 32, NQKV / 32, 1), tb, 0, stream>>>(wo, woT, NQKV, DD);

  dim3 blk(256, 1, 1);
  k_proj<<<dim3(NQKV / 64, NTOK / 128, 1), blk, 0, stream>>>(q_ip, wqT, bq, Qb, qscale, 0);
  k_proj<<<dim3(NQKV / 64, NTOK / 128, 1), blk, 0, stream>>>(k_ip, wkT, bk, Kb, 1.0f, 0);
  k_proj<<<dim3(NQKV / 64, NTOK / 128, 1), blk, 0, stream>>>(v_ip, wvT, bv, Vt, 1.0f, 1);

  k_scores<<<dim3(SS / 64, SS / 128, BB * HH), blk, 0, stream>>>(Qb, Kb, mask, attn);
  k_softmax<<<dim3(BB * HH * SS, 1, 1), dim3(256, 1, 1), 0, stream>>>(attn, Abf);
  k_ctx<<<dim3(DD / 64, SS / 128, BB * HH), blk, 0, stream>>>(Abf, Vt, Ctx);
  k_out<<<dim3(DD / 64, NTOK / 128, 1), blk, 0, stream>>>(Ctx, woT, bo, outp);
}
